// SignedGCNBlock_17540646437113
// MI455X (gfx1250) — matrix operand order repaired and checked by reading, not compiled, not run on hardware
//
#include <hip/hip_runtime.h>

#define F_IN   256
#define HCOLS  512
#define BN_EPS 1e-5f

typedef __attribute__((ext_vector_type(16))) __bf16 v16bf;
typedef __attribute__((ext_vector_type(8)))  __bf16 v8bf;
typedef __attribute__((ext_vector_type(8)))  float  v8f;

union AFrag { v16bf v; v8bf h[2]; };

// ---------------------------------------------------------------- zero fill
__global__ void k_zero(float* __restrict__ p, long long n) {
  long long i = (long long)blockIdx.x * blockDim.x + threadIdx.x;
  if (i < n) p[i] = 0.0f;
}

// ------------------------------------------------- edge scatter (mean numer.)
// one wave per edge: 32 lanes x 2 float4 = 256 floats; agg fits in 192MB L2
__global__ void __launch_bounds__(256)
k_scatter(const float* __restrict__ x, const int* __restrict__ src,
          const int* __restrict__ dst, int E,
          float* __restrict__ agg, float* __restrict__ deg) {
  int e = blockIdx.x * (blockDim.x >> 5) + (threadIdx.x >> 5);
  if (e >= E) return;
  int lane = threadIdx.x & 31;
  int s = src[e], d = dst[e];
  const float4* xs = (const float4*)(x + (size_t)s * F_IN);
  float* ag = agg + (size_t)d * F_IN;
  float4 v0 = xs[lane];
  float4 v1 = xs[lane + 32];
  int b0 = lane * 4, b1 = (lane + 32) * 4;
  unsafeAtomicAdd(ag + b0 + 0, v0.x);
  unsafeAtomicAdd(ag + b0 + 1, v0.y);
  unsafeAtomicAdd(ag + b0 + 2, v0.z);
  unsafeAtomicAdd(ag + b0 + 3, v0.w);
  unsafeAtomicAdd(ag + b1 + 0, v1.x);
  unsafeAtomicAdd(ag + b1 + 1, v1.y);
  unsafeAtomicAdd(ag + b1 + 2, v1.z);
  unsafeAtomicAdd(ag + b1 + 3, v1.w);
  if (lane == 0) unsafeAtomicAdd(deg + d, 1.0f);
}

// --------------------------------------- mean divide + f32 -> bf16 conversion
__global__ void k_meancvt(const float* __restrict__ x,
                          const float* __restrict__ aggp, const float* __restrict__ aggn,
                          const float* __restrict__ degp, const float* __restrict__ degn,
                          __bf16* __restrict__ xb, __bf16* __restrict__ mpb,
                          __bf16* __restrict__ mnb, long long total) {
  long long i = (long long)blockIdx.x * blockDim.x + threadIdx.x;
  if (i >= total) return;
  int n = (int)(i >> 8);                      // F_IN == 256
  float rp = 1.0f / fmaxf(degp[n], 1.0f);
  float rn = 1.0f / fmaxf(degn[n], 1.0f);
  xb[i]  = (__bf16)x[i];
  mpb[i] = (__bf16)(aggp[i] * rp);
  mnb[i] = (__bf16)(aggn[i] * rn);
}

// ------------- pack 4 weight matrices into transposed bf16 (B-fragment ready)
// wt[m][n][k] = W_m[k][n]  (m: 0=pos_l 1=pos_r 2=neg_l 3=neg_r)
__global__ void k_packw(const float* __restrict__ wpl, const float* __restrict__ wpr,
                        const float* __restrict__ wnl, const float* __restrict__ wnr,
                        __bf16* __restrict__ wt) {
  int i = blockIdx.x * blockDim.x + threadIdx.x;   // < 4*256*256
  int m = i >> 16;
  int r = i & 0xFFFF;
  int n = r >> 8;
  int k = r & 255;
  const float* W = (m == 0) ? wpl : (m == 1) ? wpr : (m == 2) ? wnl : wnr;
  wt[i] = (__bf16)W[k * 256 + n];
}

// --------------------------------------------------------- WMMA bf16 GEMM
// block = 8 waves, one 16-row M tile; wave w covers 4 16-col N tiles.
// waves 0-3: out_pos = meanpos@Wpl + x@Wpr + bpos ; waves 4-7: neg half.
__global__ void __launch_bounds__(256)
k_gemm(const __bf16* __restrict__ xb, const __bf16* __restrict__ mpb,
       const __bf16* __restrict__ mnb, const __bf16* __restrict__ wt,
       const float* __restrict__ bpos, const float* __restrict__ bneg,
       float* __restrict__ out, int N) {
  const int lane  = threadIdx.x & 31;
  const int wave  = threadIdx.x >> 5;        // 0..7
  const int mbase = blockIdx.x * 16;
  const int mrow  = mbase + (lane & 15);
  const int kselA = (lane & 16) ? 16 : 0;    // A fragment K base (bf16 elems)
  const int kselB = (lane & 16) ? 16 : 0;    // B fragment K base
  const int ncol  = lane & 15;

  const bool neg = wave >= 4;
  const __bf16* aggb = neg ? mnb : mpb;
  const __bf16* wl = wt + (size_t)(neg ? 2 : 0) * 256 * 256;   // [n][k]
  const __bf16* wr = wt + (size_t)(neg ? 3 : 1) * 256 * 256;
  const float*  bias = neg ? bneg : bpos;
  const int ntile0 = (wave & 3) * 4;

  const int arow = (mrow < N) ? mrow : (N - 1);
  const __bf16* arow1 = aggb + (size_t)arow * F_IN;
  const __bf16* arow2 = xb   + (size_t)arow * F_IN;

  v8f acc[4] = {};

  for (int kt = 0; kt < F_IN / 32; ++kt) {
    const int ka = kt * 32 + kselA;
    AFrag a1, a2;
    a1.h[0] = *(const v8bf*)(arow1 + ka);
    a1.h[1] = *(const v8bf*)(arow1 + ka + 8);
    a2.h[0] = *(const v8bf*)(arow2 + ka);
    a2.h[1] = *(const v8bf*)(arow2 + ka + 8);
    const int kb = kt * 32 + kselB;
#pragma unroll
    for (int j = 0; j < 4; ++j) {
      const int n = (ntile0 + j) * 16 + ncol;
      v16bf b1 = *(const v16bf*)(wl + (size_t)n * F_IN + kb);
      v16bf b2 = *(const v16bf*)(wr + (size_t)n * F_IN + kb);
      acc[j] = __builtin_amdgcn_wmma_f32_16x16x32_bf16(
          false, a1.v, false, b1, (short)0, acc[j], false, false);
      acc[j] = __builtin_amdgcn_wmma_f32_16x16x32_bf16(
          false, a2.v, false, b2, (short)0, acc[j], false, false);
    }
  }

  const int rbase = mbase + ((lane & 16) ? 8 : 0);
#pragma unroll
  for (int j = 0; j < 4; ++j) {
    const int n = (ntile0 + j) * 16 + ncol;     // 0..255 within half
    const int c = (neg ? 256 : 0) + n;
    const float bv = bias[n];
#pragma unroll
    for (int r = 0; r < 8; ++r) {
      const int row = rbase + r;
      if (row < N) out[(size_t)row * HCOLS + c] = acc[j][r] + bv;
    }
  }
}

// ------------------------------------------- per-column sum / sumsq (atomics)
__global__ void __launch_bounds__(512)
k_colstats(const float* __restrict__ h, int N, int rows_per_block,
           float* __restrict__ sums, float* __restrict__ sumsq) {
  int c  = threadIdx.x;                       // 512 columns
  int r0 = blockIdx.x * rows_per_block;
  int r1 = min(r0 + rows_per_block, N);
  float s = 0.f, q = 0.f;
  for (int r = r0; r < r1; ++r) {
    float v = h[(size_t)r * HCOLS + c];
    s += v; q += v * v;
  }
  unsafeAtomicAdd(&sums[c], s);
  unsafeAtomicAdd(&sumsq[c], q);
}

__global__ void __launch_bounds__(512)
k_bnfin(const float* __restrict__ sums, const float* __restrict__ sumsq,
        const float* __restrict__ gamma, const float* __restrict__ beta, int N,
        float* __restrict__ scale, float* __restrict__ shift) {
  int c = threadIdx.x;
  float invN = 1.0f / (float)N;
  float mu  = sums[c] * invN;
  float var = fmaxf(sumsq[c] * invN - mu * mu, 0.0f);
  float sc  = gamma[c] * rsqrtf(var + BN_EPS);
  scale[c] = sc;
  shift[c] = beta[c] - mu * sc;
}

__global__ void k_bnapply(float* __restrict__ h, long long total,
                          const float* __restrict__ scale,
                          const float* __restrict__ shift) {
  long long i = (long long)blockIdx.x * blockDim.x + threadIdx.x;
  if (i >= total) return;
  int c = (int)(i & (HCOLS - 1));
  h[i] = fmaxf(fmaf(h[i], scale[c], shift[c]), 0.0f);
}

// ---------------------------------------------------------------------------
extern "C" void kernel_launch(void* const* d_in, const int* in_sizes, int n_in,
                              void* d_out, int out_size, void* d_ws, size_t ws_size,
                              hipStream_t stream) {
  const float* x    = (const float*)d_in[0];
  const int*   pei  = (const int*)d_in[1];
  const int*   nei  = (const int*)d_in[2];
  const float* wpl  = (const float*)d_in[3];
  const float* wpr  = (const float*)d_in[4];
  const float* bpos = (const float*)d_in[5];
  const float* wnl  = (const float*)d_in[6];
  const float* wnr  = (const float*)d_in[7];
  const float* bneg = (const float*)d_in[8];
  const float* gamma= (const float*)d_in[9];
  const float* beta = (const float*)d_in[10];
  const int N = in_sizes[0] / F_IN;
  const int E = in_sizes[1] / 2;
  float* out = (float*)d_out;

  // ---- workspace carve-out (aligned 256B)
  char* ws = (char*)d_ws;
  size_t off = 0;
  auto alloc = [&](size_t bytes) -> char* {
    off = (off + 255) & ~(size_t)255;
    char* p = ws + off;
    off += bytes;
    return p;
  };
  float*  aggp  = (float*)alloc((size_t)N * F_IN * 4);
  float*  aggn  = (float*)alloc((size_t)N * F_IN * 4);
  float*  degp  = (float*)alloc((size_t)N * 4);
  float*  degn  = (float*)alloc((size_t)N * 4);
  float*  sums  = (float*)alloc(HCOLS * 4);
  float*  sumsq = (float*)alloc(HCOLS * 4);
  float*  scale = (float*)alloc(HCOLS * 4);
  float*  shift = (float*)alloc(HCOLS * 4);
  __bf16* xb    = (__bf16*)alloc((size_t)N * F_IN * 2);
  __bf16* mpb   = (__bf16*)alloc((size_t)N * F_IN * 2);
  __bf16* mnb   = (__bf16*)alloc((size_t)N * F_IN * 2);
  __bf16* wt    = (__bf16*)alloc((size_t)4 * 256 * 256 * 2);

  // 1) zero accumulation region (aggp .. sumsq, contiguous incl. padding)
  {
    long long cnt = (long long)(((char*)sumsq + HCOLS * 4) - (char*)aggp) / 4;
    int grid = (int)((cnt + 255) / 256);
    k_zero<<<grid, 256, 0, stream>>>(aggp, cnt);
  }
  // 2) edge scatters (pos, neg): 8 edges / 256-thread block
  {
    int grid = (E + 7) / 8;
    k_scatter<<<grid, 256, 0, stream>>>(x, pei, pei + E, E, aggp, degp);
    k_scatter<<<grid, 256, 0, stream>>>(x, nei, nei + E, E, aggn, degn);
  }
  // 3) mean + bf16 conversion
  {
    long long tot = (long long)N * F_IN;
    int grid = (int)((tot + 255) / 256);
    k_meancvt<<<grid, 256, 0, stream>>>(x, aggp, aggn, degp, degn, xb, mpb, mnb, tot);
  }
  // 4) pack weights (transposed bf16)
  k_packw<<<(4 * 256 * 256) / 256, 256, 0, stream>>>(wpl, wpr, wnl, wnr, wt);
  // 5) WMMA GEMM + bias -> pre-BN h into d_out
  k_gemm<<<(N + 15) / 16, 256, 0, stream>>>(xb, mpb, mnb, wt, bpos, bneg, out, N);
  // 6) column stats
  {
    const int rpb = 512;
    int grid = (N + rpb - 1) / rpb;
    k_colstats<<<grid, 512, 0, stream>>>(out, N, rpb, sums, sumsq);
  }
  // 7) BN finalize + apply (ReLU fused), in place on d_out
  k_bnfin<<<1, 512, 0, stream>>>(sums, sumsq, gamma, beta, N, scale, shift);
  {
    long long tot = (long long)N * HCOLS;
    int grid = (int)((tot + 255) / 256);
    k_bnapply<<<grid, 256, 0, stream>>>(out, tot, scale, shift);
  }
}
